// RoPEAttentionHead_81217831567434
// MI455X (gfx1250) — compile-verified, matrix operand order checked by reading, not run on hardware
//
#include <hip/hip_runtime.h>
#include <hip/hip_bf16.h>
#include <stdint.h>

#define T_SEQ 2048
#define DMODEL 1024
#define DH 128
#define BATCH 4
#define NSPLIT 4   // KV-split factor for the attention kernel (flash-decoding style)

typedef __attribute__((ext_vector_type(16))) __bf16 bf16x16;
typedef __attribute__((ext_vector_type(8))) float f32x8;

union FragBF {
  bf16x16 v;
  unsigned short u16[16];
  uint4 q[2];
};
union Acc8 {
  f32x8 v;
  float f[8];
};

// Packed f32x2 -> bf16x2 (RNE); lowers to v_cvt_pk_bf16_f32 on gfx1250.
static __device__ __forceinline__ unsigned int pack2bf(float a, float b) {
  __hip_bfloat162 h = __float22bfloat162_rn(make_float2(a, b));
  unsigned int u;
  __builtin_memcpy(&u, &h, 4);
  return u;
}

static __device__ __forceinline__ unsigned short f2bf(float f) {
  unsigned int u = __float_as_uint(f);
  u += 0x7FFFu + ((u >> 16) & 1u);   // round-to-nearest-even
  return (unsigned short)(u >> 16);
}

static __device__ __forceinline__ f32x8 wmma_bf16(const FragBF& a, const FragBF& b, f32x8 c) {
  // D = A(16x32 bf16) * B(32x16 bf16) + C(16x16 f32)
  return __builtin_amdgcn_wmma_f32_16x16x32_bf16(false, a.v, false, b.v, (short)0, c, false, false);
}

// ---------------------------------------------------------------------------
// Kernel 1: Q/K/V projection (WMMA bf16) + RoPE on Q,K (+1/sqrt(D) folded into Q)
// grid = (8192/64, 3), block = 128 (4 waves). Wave w: M rows [m0+16w, +16), all 128 N.
// Outputs: Qb,Kb row-major [B*T,128] bf16 ; V transposed Vt [B][128][T] bf16.
// ---------------------------------------------------------------------------
__global__ __launch_bounds__(128) void proj_rope_kernel(
    const float* __restrict__ x, const float* __restrict__ Wq,
    const float* __restrict__ Wk, const float* __restrict__ Wv,
    const float* __restrict__ theta,
    unsigned short* __restrict__ Qb, unsigned short* __restrict__ Kb,
    unsigned short* __restrict__ Vt) {
  __shared__ __align__(16) unsigned short lx[64 * 32];    // x tile (bf16)
  __shared__ __align__(16) unsigned short lw[128 * 32];   // W tile (bf16)

  const int which = blockIdx.y;
  const float* W = (which == 0) ? Wq : (which == 1) ? Wk : Wv;
  const int m0 = blockIdx.x * 64;
  const int tid = threadIdx.x;
  const int lane = tid & 31;
  const int wv = tid >> 5;
  const int half = lane >> 4;
  const int lr = lane & 15;

  Acc8 acc[8];
#pragma unroll
  for (int nt = 0; nt < 8; ++nt)
#pragma unroll
    for (int r = 0; r < 8; ++r) acc[nt].f[r] = 0.f;

  for (int kc = 0; kc < DMODEL / 32; ++kc) {
    // cooperative fp32->bf16 staging: x tile 64x32 (512 float4), W tile 128x32 (1024 float4)
#pragma unroll
    for (int j = 0; j < 4; ++j) {
      int i = tid + j * 128;
      int row = i >> 3, c4 = i & 7;
      float4 f = *(const float4*)(x + (size_t)(m0 + row) * DMODEL + kc * 32 + c4 * 4);
      uint2 p;
      p.x = pack2bf(f.x, f.y);
      p.y = pack2bf(f.z, f.w);
      *(uint2*)(&lx[row * 32 + c4 * 4]) = p;
    }
#pragma unroll
    for (int j = 0; j < 8; ++j) {
      int i = tid + j * 128;
      int row = i >> 3, c4 = i & 7;
      float4 f = *(const float4*)(W + (size_t)row * DMODEL + kc * 32 + c4 * 4);
      uint2 p;
      p.x = pack2bf(f.x, f.y);
      p.y = pack2bf(f.z, f.w);
      *(uint2*)(&lw[row * 32 + c4 * 4]) = p;
    }
    __syncthreads();

    // A fragment: x rows, K in {8h..8h+7} ∪ {16+8h..16+8h+7} per lane -> two b128
    FragBF A;
    const unsigned short* arow = &lx[(16 * wv + lr) * 32];
    A.q[0] = *(const uint4*)(arow + 8 * half);
    A.q[1] = *(const uint4*)(arow + 16 + 8 * half);

#pragma unroll
    for (int nt = 0; nt < 8; ++nt) {
      // B fragment: B[k][n] = W[n][k]; lane n=lr reads W row contiguously along d
      FragBF Bf;
      const unsigned short* brow = &lw[(16 * nt + lr) * 32];
      Bf.q[0] = *(const uint4*)(brow + 8 * half);
      Bf.q[1] = *(const uint4*)(brow + 16 + 8 * half);
      acc[nt].v = wmma_bf16(A, Bf, acc[nt].v);
    }
    __syncthreads();
  }

  // Epilogue. C layout: element (nt, r, lane): M = r + 8*half (token row), N = 16*nt + lr (head dim)
  if (which < 2) {
    unsigned short* dst = (which == 0) ? Qb : Kb;
    const float qscale = (which == 0) ? 0.03125f : 1.0f;  // 1/sqrt(1024) folded into Q
#pragma unroll
    for (int nt = 0; nt < 8; ++nt) {
      int n = 16 * nt + lr;
      float th = theta[n];
      float sgn = (n & 1) ? -1.f : 1.f;   // (x0,x1) -> (x0 c + x1 s, x1 c - x0 s)
#pragma unroll
      for (int r = 0; r < 8; ++r) {
        int m = m0 + 16 * wv + r + 8 * half;
        int t = m & (T_SEQ - 1);
        float ang = (float)(t + 1) * th;
        float s, c;
        __sincosf(ang, &s, &c);
        float xv = acc[nt].f[r];
        float xp = __shfl_xor(xv, 1, 32);   // pair partner (adjacent head dim = adjacent lane)
        float o = (xv * c + sgn * xp * s) * qscale;
        dst[(size_t)m * DH + n] = f2bf(o);
      }
    }
  } else {
    // V: store transposed Vt[b][dh][t]. Consecutive r = consecutive t -> one b128 per tile.
    int mb = m0 + 16 * wv + 8 * half;   // base token of this lane's 8 accumulator rows
    int b = mb >> 11;
    int t0 = mb & (T_SEQ - 1);
#pragma unroll
    for (int nt = 0; nt < 8; ++nt) {
      int n = 16 * nt + lr;
      uint4 p;
      p.x = pack2bf(acc[nt].f[0], acc[nt].f[1]);
      p.y = pack2bf(acc[nt].f[2], acc[nt].f[3]);
      p.z = pack2bf(acc[nt].f[4], acc[nt].f[5]);
      p.w = pack2bf(acc[nt].f[6], acc[nt].f[7]);
      *(uint4*)(&Vt[((size_t)b * DH + n) * T_SEQ + t0]) = p;
    }
  }
}

// ---------------------------------------------------------------------------
// Kernel 2: causal flash attention with 4-way KV split, one wave per (16 q-rows, split).
// Computes S^T = K·Q^T (q-columns per-lane), then O^T += V^T·P^T.
// C layout of S^T coincides with B-operand layout of P^T -> zero cross-lane P movement.
// grid = (T/16, B, NSPLIT), block = 32 (1 wave). Partials (unnormalized O, m, l) -> ws.
// ---------------------------------------------------------------------------
__global__ __launch_bounds__(32) void attn_kernel(
    const unsigned short* __restrict__ Qb, const unsigned short* __restrict__ Kb,
    const unsigned short* __restrict__ Vt,
    float* __restrict__ Opart, float* __restrict__ Mpart, float* __restrict__ Lpart) {
  const int lane = threadIdx.x & 31;
  const int half = lane >> 4;
  const int lr = lane & 15;
  const int b = blockIdx.y;
  const int split = blockIdx.z;
  const int qbase = blockIdx.x * 16;
  const int qend = qbase + 15;
  const int q = qbase + lr;   // this lane's q column (both halves hold same q)

  const unsigned short* Qp = Qb + (size_t)b * T_SEQ * DH;
  const unsigned short* Kp = Kb + (size_t)b * T_SEQ * DH;
  const unsigned short* Vp = Vt + (size_t)b * DH * T_SEQ;

  // Q^T B-operand fragments, one per 32-wide d-chunk (lane holds its full Q row)
  FragBF Bq[4];
#pragma unroll
  for (int dc = 0; dc < 4; ++dc) {
    const unsigned short* qrow = Qp + (size_t)q * DH + dc * 32;
    Bq[dc].q[0] = *(const uint4*)(qrow + 8 * half);
    Bq[dc].q[1] = *(const uint4*)(qrow + 16 + 8 * half);
  }

  Acc8 O[8];          // O^T in C layout: M = head dim (8 tiles of 16), N = q
#pragma unroll
  for (int dt = 0; dt < 8; ++dt)
#pragma unroll
    for (int r = 0; r < 8; ++r) O[dt].f[r] = 0.f;
  float mrun = -1e30f, lrun = 0.f;

  // Round-robin KV block assignment across splits balances the causal skew.
  for (int j0 = 32 * split; j0 <= qend; j0 += 32 * NSPLIT) {
    {  // prefetch next K block for this split (uniform branch; speculative)
      int jn = j0 + 32 * NSPLIT;
      if (jn <= qend) __builtin_prefetch(Kp + (size_t)(jn + lr) * DH, 0, 3);
    }
    Acc8 S0, S1;   // S^T tiles: kv rows [j0,j0+16) and [j0+16,j0+32)
#pragma unroll
    for (int r = 0; r < 8; ++r) { S0.f[r] = 0.f; S1.f[r] = 0.f; }
#pragma unroll
    for (int dc = 0; dc < 4; ++dc) {
      FragBF K0, K1;
      const unsigned short* k0r = Kp + (size_t)(j0 + lr) * DH + dc * 32;
      K0.q[0] = *(const uint4*)(k0r + 8 * half);
      K0.q[1] = *(const uint4*)(k0r + 16 + 8 * half);
      const unsigned short* k1r = Kp + (size_t)(j0 + 16 + lr) * DH + dc * 32;
      K1.q[0] = *(const uint4*)(k1r + 8 * half);
      K1.q[1] = *(const uint4*)(k1r + 16 + 8 * half);
      S0.v = wmma_bf16(K0, Bq[dc], S0.v);
      S1.v = wmma_bf16(K1, Bq[dc], S1.v);
    }

    // causal mask + per-column (per-lane) max
    float bm = -1e30f;
#pragma unroll
    for (int r = 0; r < 8; ++r) {
      int kv0 = j0 + r + 8 * half;
      if (kv0 > q) S0.f[r] = -1e30f;
      if (kv0 + 16 > q) S1.f[r] = -1e30f;
      bm = fmaxf(bm, fmaxf(S0.f[r], S1.f[r]));
    }
    bm = fmaxf(bm, __shfl_xor(bm, 16, 32));
    float mn = fmaxf(mrun, bm);
    float alpha = __expf(mrun - mn);

    // P = exp(S - m); pack P^T directly into B-operand layout (pure in-lane)
    float rl = 0.f;
    FragBF Pf;
#pragma unroll
    for (int r = 0; r < 8; ++r) {
      float e0 = __expf(S0.f[r] - mn);
      float e1 = __expf(S1.f[r] - mn);
      rl += e0 + e1;
      Pf.u16[r] = f2bf(e0);       // element e<8  <-> kv = j0 + 8h + e
      Pf.u16[8 + r] = f2bf(e1);   // element e>=8 <-> kv = j0 + 16 + 8h + (e-8)
    }
    rl += __shfl_xor(rl, 16, 32);
    lrun = lrun * alpha + rl;
    mrun = mn;

#pragma unroll
    for (int dt = 0; dt < 8; ++dt) O[dt].v *= alpha;

    // O^T += V^T · P^T ; A = V^T tile from transposed V (contiguous along t)
#pragma unroll
    for (int dt = 0; dt < 8; ++dt) {
      FragBF Vf;
      const unsigned short* vrow = Vp + (size_t)(16 * dt + lr) * T_SEQ + j0;
      Vf.q[0] = *(const uint4*)(vrow + 8 * half);
      Vf.q[1] = *(const uint4*)(vrow + 16 + 8 * half);
      O[dt].v = wmma_bf16(Vf, Pf, O[dt].v);
    }
  }

  // Write unnormalized partials (zero iterations -> O=0, m=-1e30, l=0: valid identity).
  const size_t prow = ((size_t)split * BATCH + b) * T_SEQ + q;
  float* Op = Opart + prow * DH;
#pragma unroll
  for (int dt = 0; dt < 8; ++dt)
#pragma unroll
    for (int r = 0; r < 8; ++r) {
      int dh = 16 * dt + r + 8 * half;
      Op[dh] = O[dt].f[r];
    }
  Mpart[prow] = mrun;   // both halves write the same value: benign
  Lpart[prow] = lrun;
}

// ---------------------------------------------------------------------------
// Kernel 3: merge KV-split partials. out = sum_s exp(m_s - M) O_s / sum_s exp(m_s - M) l_s
// grid = B*T*DH/256, block = 256. L2-resident traffic (~20 MB).
// ---------------------------------------------------------------------------
__global__ __launch_bounds__(256) void reduce_kernel(
    const float* __restrict__ Opart, const float* __restrict__ Mpart,
    const float* __restrict__ Lpart, float* __restrict__ out) {
  size_t idx = (size_t)blockIdx.x * 256 + threadIdx.x;   // over B*T*DH
  size_t row = idx >> 7;                                  // b*T + t
  int dh = (int)(idx & (DH - 1));

  float m[NSPLIT];
  float M = -1e30f;
#pragma unroll
  for (int s = 0; s < NSPLIT; ++s) {
    m[s] = Mpart[(size_t)s * (BATCH * T_SEQ) + row];
    M = fmaxf(M, m[s]);
  }
  float denom = 0.f, num = 0.f;
#pragma unroll
  for (int s = 0; s < NSPLIT; ++s) {
    float w = __expf(m[s] - M);
    denom += w * Lpart[(size_t)s * (BATCH * T_SEQ) + row];
    num += w * Opart[((size_t)s * (BATCH * T_SEQ) + row) * DH + dh];
  }
  out[row * DH + dh] = num / denom;
}

extern "C" void kernel_launch(void* const* d_in, const int* in_sizes, int n_in,
                              void* d_out, int out_size, void* d_ws, size_t ws_size,
                              hipStream_t stream) {
  (void)in_sizes; (void)n_in; (void)out_size; (void)ws_size;
  const float* x = (const float*)d_in[0];
  const float* Wq = (const float*)d_in[1];
  const float* Wk = (const float*)d_in[2];
  const float* Wv = (const float*)d_in[3];
  const float* theta = (const float*)d_in[4];
  float* out = (float*)d_out;

  // workspace layout:
  //   Qb | Kb : row-major [B*T,128] bf16 (2 MB each)
  //   Vt      : [B][128][T] bf16 (2 MB)
  //   Opart   : [NSPLIT][B*T][128] f32 (16 MB)
  //   Mpart, Lpart : [NSPLIT][B*T] f32 (128 KB each)
  unsigned short* Qb = (unsigned short*)d_ws;
  unsigned short* Kb = Qb + (size_t)BATCH * T_SEQ * DH;
  unsigned short* Vt = Kb + (size_t)BATCH * T_SEQ * DH;
  float* Opart = (float*)(Vt + (size_t)BATCH * T_SEQ * DH);
  float* Mpart = Opart + (size_t)NSPLIT * BATCH * T_SEQ * DH;
  float* Lpart = Mpart + (size_t)NSPLIT * BATCH * T_SEQ;

  dim3 g1(BATCH * T_SEQ / 64, 3);
  proj_rope_kernel<<<g1, 128, 0, stream>>>(x, Wq, Wk, Wv, theta, Qb, Kb, Vt);

  dim3 g2(T_SEQ / 16, BATCH, NSPLIT);
  attn_kernel<<<g2, 32, 0, stream>>>(Qb, Kb, Vt, Opart, Mpart, Lpart);

  dim3 g3((BATCH * T_SEQ * DH) / 256);
  reduce_kernel<<<g3, 256, 0, stream>>>(Opart, Mpart, Lpart, out);
}
